// AttentionBlock_80968723464882
// MI455X (gfx1250) — matrix operand order repaired and checked by reading, not compiled, not run on hardware
//
#include <hip/hip_runtime.h>

// ---------------------------------------------------------------------------
// Types for CDNA5 WMMA (wave32, gfx1250)
// ---------------------------------------------------------------------------
typedef __attribute__((ext_vector_type(16))) __bf16 v16bf;
typedef __attribute__((ext_vector_type(8)))  float  v8f;

union AFrag { v16bf v; unsigned int d[8]; };

__device__ inline v8f wmma_bf16(v16bf a, v16bf b, v8f c) {
  // D = A(16x32) * B(32x16) + C(16x16), fp32 accumulate
  return __builtin_amdgcn_wmma_f32_16x16x32_bf16(false, a, false, b,
                                                 (short)0, c, false, false);
}

// Async DMA: per-lane 16B global -> LDS, tracked by ASYNCcnt (CDNA5).
__device__ inline void async_b128(void* lds_dst, const void* gsrc) {
  unsigned ldsa = (unsigned)(uintptr_t)lds_dst;          // low 32b = LDS offset
  unsigned long long ga = (unsigned long long)(uintptr_t)gsrc;
  asm volatile("global_load_async_to_lds_b128 %0, %1, off"
               :: "v"(ldsa), "v"(ga) : "memory");
}
#define WAIT_ASYNC(n) asm volatile("s_wait_asynccnt " #n ::: "memory")

__device__ inline unsigned short f2bf(float f) {
  unsigned int u = __float_as_uint(f);
  unsigned int r = u + 0x7FFFu + ((u >> 16) & 1u);  // round-to-nearest-even
  return (unsigned short)(r >> 16);
}

__device__ inline float half_reduce_max(float v) {
#pragma unroll
  for (int off = 8; off > 0; off >>= 1) v = fmaxf(v, __shfl_xor(v, off, 32));
  return v;  // reduces across the 16 lanes of a half-wave
}
__device__ inline float half_reduce_sum(float v) {
#pragma unroll
  for (int off = 8; off > 0; off >>= 1) v += __shfl_xor(v, off, 32);
  return v;
}

// Problem constants
#define BB   8
#define LL   2048
#define EE   1024
#define NDD  512
#define FFD  2048
#define BL   16384            // B*L
#define QK_SCALE 0.476538688f // sqrt(1/1024) * 2 * ln(2048)

// ---------------------------------------------------------------------------
// fp32 -> bf16 bulk convert (vectorized x4)
// ---------------------------------------------------------------------------
__global__ void f32_to_bf16_k(const float* __restrict__ s,
                              unsigned short* __restrict__ d, long n4) {
  long i = (long)blockIdx.x * blockDim.x + threadIdx.x;
  if (i < n4) {
    float4 f = ((const float4*)s)[i];
    ushort4 u;
    u.x = f2bf(f.x); u.y = f2bf(f.y); u.z = f2bf(f.z); u.w = f2bf(f.w);
    ((ushort4*)d)[i] = u;
  }
}

// ---------------------------------------------------------------------------
// Generic tiled WMMA GEMM:  C[m,n] = sum_k A[m,k] * Bw[n,k]
//   A, Bw bf16 row-major (K contiguous). WG tile 64(M) x 128(N), 8 waves,
//   each wave owns 16x64. K staged 64 cols at a time via async-to-LDS DMA,
//   double-buffered so DMA overlaps WMMA.
// EPI: 0 = QKV split (q scaled, v transposed), 1 = FF1 (+b1, relu, bf16),
//      2 = FF2 (+b2 + residual, fp32)
// ---------------------------------------------------------------------------
template <int KD, int EPI>
__launch_bounds__(256, 2)
__global__ void gemm_bf16_wmma(const unsigned short* __restrict__ A,
                               const unsigned short* __restrict__ Bw,
                               const float* __restrict__ bias,
                               const float* __restrict__ resid,
                               float* __restrict__ outF,
                               unsigned short* __restrict__ outB0,
                               unsigned short* __restrict__ outB1,
                               unsigned short* __restrict__ outB2) {
  __shared__ __align__(32) unsigned short Asm[2][64 * 64];    // 2 x 8 KB
  __shared__ __align__(32) unsigned short Bsm[2][128 * 64];   // 2 x 16 KB

  const int tid  = threadIdx.x;
  const int wave = tid >> 5, lane = tid & 31;
  const int g = lane >> 4, hl = lane & 15;
  const int wm = wave >> 1, wn = wave & 1;
  const long m0 = (long)blockIdx.x * 64;
  const long n0 = (long)blockIdx.y * 128;

  v8f acc[4] = {};

  // Async-stage one K tile (64 cols) into buffer `b`.
  auto stage = [&](int kn, int b) {
#pragma unroll
    for (int ch = 0; ch < 2; ++ch) {
      int i = tid + ch * 256;
      int r = i >> 3, c = i & 7;
      async_b128(&Asm[b][r * 64 + c * 8], A + (m0 + r) * KD + kn + c * 8);
    }
#pragma unroll
    for (int ch = 0; ch < 4; ++ch) {
      int i = tid + ch * 256;
      int r = i >> 3, c = i & 7;
      async_b128(&Bsm[b][r * 64 + c * 8], Bw + (n0 + r) * KD + kn + c * 8);
    }
  };

  stage(0, 0);
  WAIT_ASYNC(0);
  __syncthreads();

  for (int k0 = 0; k0 < KD; k0 += 64) {
    const int b = (k0 >> 6) & 1;
    if (k0 + 64 < KD) stage(k0 + 64, b ^ 1);  // DMA next tile during compute

#pragma unroll
    for (int kk = 0; kk < 64; kk += 32) {
      // A fragment: lanes 0-15 -> K {0..15}, lanes 16-31 -> {16..31}
      AFrag af;
#pragma unroll
      for (int dd = 0; dd < 8; ++dd) {
        int K0 = kk + 2 * dd + 16 * g;
        af.d[dd] = ((const unsigned int*)Asm[b])[(wm * 16 + hl) * 32 + (K0 >> 1)];
      }
      v16bf bf[4];
#pragma unroll
      for (int t = 0; t < 4; ++t)
        bf[t] = *(const v16bf*)(Bsm[b] +
                                ((wn * 64 + t * 16 + hl) * 64 + kk + g * 16));
#pragma unroll
      for (int t = 0; t < 4; ++t) acc[t] = wmma_bf16(af.v, bf[t], acc[t]);
    }
    WAIT_ASYNC(0);
    __syncthreads();
  }

  // Epilogue: C VGPR j -> row (j | j+8), lane -> col
#pragma unroll
  for (int t = 0; t < 4; ++t) {
#pragma unroll
    for (int j = 0; j < 8; ++j) {
      long row = m0 + wm * 16 + j + 8 * g;
      long col = n0 + wn * 64 + t * 16 + hl;
      float v = acc[t][j];
      if (EPI == 0) {
        if (col < 1024) {
          outB0[row * 1024 + col] = f2bf(v * QK_SCALE);          // q (pre-scaled)
        } else if (col < 2048) {
          outB1[row * 1024 + (col - 1024)] = f2bf(v);            // k
        } else {
          long dd = col - 2048;                                   // vT[b][d][s]
          long b2 = row >> 11, s = row & 2047;
          outB2[((b2 << 9) + dd) * 2048 + s] = f2bf(v);
        }
      } else if (EPI == 1) {
        v += bias[col];
        v = fmaxf(v, 0.0f);
        outB0[row * (long)FFD + col] = f2bf(v);
      } else {
        v += bias[col] + resid[row * (long)NDD + col];
        outF[row * (long)NDD + col] = v;
      }
    }
  }
}

// ---------------------------------------------------------------------------
// Flash attention: 4 waves x 16 rows = 64 rows of one batch per block.
// LDS: Q tile 64x1024 (128 KB, loaded once) + K double buffer (2 x 32x1024,
// 128 KB) + vT tile 512x32 (32 KB) + P scratch (4 KB) = 292 KB of WGP LDS.
// All staging via async-to-LDS DMA; K(next) and V(cur) in flight during S.
// ---------------------------------------------------------------------------
#define FA_OFF_Q 0
#define FA_OFF_K 131072
#define FA_OFF_V 262144
#define FA_OFF_P 294912
#define FA_LDS_BYTES 299008

__launch_bounds__(128, 1)
__global__ void flash_attn_k(const unsigned short* __restrict__ qg,
                             const unsigned short* __restrict__ kg,
                             const unsigned short* __restrict__ vTg,
                             float* __restrict__ o) {
  extern __shared__ __align__(32) unsigned char smem[];
  unsigned short* Qlds = (unsigned short*)(smem + FA_OFF_Q);   // [64][1024]
  unsigned short* Klds = (unsigned short*)(smem + FA_OFF_K);   // 2 x [32][1024]
  unsigned short* Vlds = (unsigned short*)(smem + FA_OFF_V);   // [512][32]
  const int tid = threadIdx.x;
  const int wave = tid >> 5, lane = tid & 31;
  const int g = lane >> 4, hl = lane & 15;
  unsigned short* Pw = (unsigned short*)(smem + FA_OFF_P) + wave * 512;

  const int bb = blockIdx.x >> 5;   // batch
  const int mt = blockIdx.x & 31;   // 64-row tile within batch
  const long brow = (long)bb * LL + mt * 64 + wave * 16;
  const int qrow = wave * 16 + hl;  // this lane's A-fragment row in Qlds

  auto stage_k = [&](int s0, unsigned short* Kb) {  // 32 async instr / wave
    for (int i = tid; i < 4096; i += 128) {
      int r = i >> 7, c = i & 127;
      async_b128(Kb + r * 1024 + c * 8,
                 kg + ((long)bb * LL + s0 + r) * EE + c * 8);
    }
  };
  auto stage_v = [&](int s0) {                      // 16 async instr / wave
    for (int i = tid; i < 2048; i += 128) {
      int r = i >> 2, c = i & 3;
      async_b128(Vlds + r * 32 + c * 8,
                 vTg + ((long)bb * NDD + r) * (long)LL + s0 + c * 8);
    }
  };

  // Prologue: Q tile (once) + first K tile.
  for (int i = tid; i < 8192; i += 128) {
    int r = i >> 7, c = i & 127;
    async_b128(Qlds + r * 1024 + c * 8,
               qg + ((long)bb * LL + mt * 64 + r) * EE + c * 8);
  }
  stage_k(0, Klds);
  WAIT_ASYNC(0);
  __syncthreads();

  v8f accO[32] = {};
  float m_r[8], l_r[8];
#pragma unroll
  for (int j = 0; j < 8; ++j) { m_r[j] = -3.4e38f; l_r[j] = 0.0f; }

  for (int it = 0; it < LL / 32; ++it) {
    const int s0 = it * 32;
    const unsigned short* Kb = Klds + (it & 1) * 32768;
    const bool more = (it + 1) < (LL / 32);

    // Issue V(cur) then K(next): V completes first (in-order ASYNCcnt).
    stage_v(s0);
    if (more) stage_k(s0 + 32, Klds + ((it + 1) & 1) * 32768);

    // S = Q_tile (16x1024) @ K_tile^T -> 16x32 (two N-tiles), pipelined frags.
    v8f sA = {}, sB = {};
    struct { AFrag a; v16bf b0, b1; } fc, fn;
    {
#pragma unroll
      for (int dd = 0; dd < 8; ++dd) {
        int K0 = 2 * dd + 16 * g;
        fc.a.d[dd] = ((const unsigned int*)Qlds)[qrow * 512 + (K0 >> 1)];
      }
      fc.b0 = *(const v16bf*)(Kb + hl * 1024 + g * 16);
      fc.b1 = *(const v16bf*)(Kb + (16 + hl) * 1024 + g * 16);
    }
    fn = fc;
    for (int kk = 0; kk < EE; kk += 32) {
      if (kk + 32 < EE) {
        int kn = kk + 32;
#pragma unroll
        for (int dd = 0; dd < 8; ++dd) {
          int K0 = kn + 2 * dd + 16 * g;
          fn.a.d[dd] = ((const unsigned int*)Qlds)[qrow * 512 + (K0 >> 1)];
        }
        fn.b0 = *(const v16bf*)(Kb + hl * 1024 + kn + g * 16);
        fn.b1 = *(const v16bf*)(Kb + (16 + hl) * 1024 + kn + g * 16);
      }
      sA = wmma_bf16(fc.a.v, fc.b0, sA);
      sB = wmma_bf16(fc.a.v, fc.b1, sB);
      fc = fn;
    }

    // Online softmax per row (row = j + 8g on this half-wave)
#pragma unroll
    for (int j = 0; j < 8; ++j) {
      float a = sA[j], b = sB[j];
      float mx = half_reduce_max(fmaxf(a, b));
      float mnew = fmaxf(m_r[j], mx);
      float corr = __expf(m_r[j] - mnew);
      float p0 = __expf(a - mnew);
      float p1 = __expf(b - mnew);
      float ps = half_reduce_sum(p0 + p1);
      l_r[j] = l_r[j] * corr + ps;
      m_r[j] = mnew;
      Pw[(j + 8 * g) * 32 + hl]      = f2bf(p0);
      Pw[(j + 8 * g) * 32 + 16 + hl] = f2bf(p1);
#pragma unroll
      for (int t = 0; t < 32; ++t) accO[t][j] *= corr;
    }

    // V(cur) is ready when only K(next)'s 32 async ops remain in flight.
    if (more) { WAIT_ASYNC(32); } else { WAIT_ASYNC(0); }
    __syncthreads();

    // Re-read P in A-fragment layout (same-wave LDS, DS ops are in-order)
    AFrag pf;
#pragma unroll
    for (int dd = 0; dd < 8; ++dd) {
      int K0 = 2 * dd + 16 * g;
      unsigned int lo = Pw[hl * 32 + K0];
      unsigned int hi = Pw[hl * 32 + K0 + 1];
      pf.d[dd] = lo | (hi << 16);
    }

    // O += P (16x32) @ vT_tile^T  (k-dim = 32 columns of this s-tile)
#pragma unroll
    for (int t = 0; t < 32; ++t) {
      const v16bf* bv = (const v16bf*)(Vlds + (t * 16 + hl) * 32 + g * 16);
      accO[t] = wmma_bf16(pf.v, *bv, accO[t]);
    }

    WAIT_ASYNC(0);     // K(next) landed; all waves done reading this K buffer
    __syncthreads();
  }

  float inv[8];
#pragma unroll
  for (int j = 0; j < 8; ++j) inv[j] = 1.0f / l_r[j];
#pragma unroll
  for (int t = 0; t < 32; ++t)
#pragma unroll
    for (int j = 0; j < 8; ++j)
      o[(brow + j + 8 * g) * (long)NDD + t * 16 + hl] = accO[t][j] * inv[j];
}

// ---------------------------------------------------------------------------
// LayerNorm over 512-wide rows: one wave per row, 8 rows per block.
// ---------------------------------------------------------------------------
__launch_bounds__(256)
__global__ void ln512_k(const float* __restrict__ in,
                        const float* __restrict__ gm,
                        const float* __restrict__ bt,
                        float* __restrict__ outF,
                        unsigned short* __restrict__ outB) {
  const int wave = threadIdx.x >> 5, lane = threadIdx.x & 31;
  const long row = (long)blockIdx.x * 8 + wave;
  const float4* r4 = (const float4*)(in + row * NDD);
  float4 x[4];
#pragma unroll
  for (int c = 0; c < 4; ++c) x[c] = r4[lane + 32 * c];

  float s = 0.f, q = 0.f;
#pragma unroll
  for (int c = 0; c < 4; ++c) {
    s += x[c].x + x[c].y + x[c].z + x[c].w;
    q += x[c].x * x[c].x + x[c].y * x[c].y + x[c].z * x[c].z + x[c].w * x[c].w;
  }
#pragma unroll
  for (int off = 16; off > 0; off >>= 1) {
    s += __shfl_xor(s, off, 32);
    q += __shfl_xor(q, off, 32);
  }
  float mean = s * (1.0f / NDD);
  float var  = q * (1.0f / NDD) - mean * mean;
  float rstd = rsqrtf(var + 1e-5f);

#pragma unroll
  for (int c = 0; c < 4; ++c) {
    int v4 = lane + 32 * c;
    float4 gv = ((const float4*)gm)[v4];
    float4 bv = ((const float4*)bt)[v4];
    float4 y;
    y.x = (x[c].x - mean) * rstd * gv.x + bv.x;
    y.y = (x[c].y - mean) * rstd * gv.y + bv.y;
    y.z = (x[c].z - mean) * rstd * gv.z + bv.z;
    y.w = (x[c].w - mean) * rstd * gv.w + bv.w;
    if (outF) ((float4*)(outF + row * NDD))[v4] = y;
    if (outB) {
      ushort4 u;
      u.x = f2bf(y.x); u.y = f2bf(y.y); u.z = f2bf(y.z); u.w = f2bf(y.w);
      ((ushort4*)(outB + row * NDD))[v4] = u;
    }
  }
}

// ---------------------------------------------------------------------------
// Orchestration
// ---------------------------------------------------------------------------
extern "C" void kernel_launch(void* const* d_in, const int* in_sizes, int n_in,
                              void* d_out, int out_size, void* d_ws,
                              size_t ws_size, hipStream_t stream) {
  (void)in_sizes; (void)n_in; (void)out_size; (void)ws_size;

  const float* x   = (const float*)d_in[0];
  const float* wp  = (const float*)d_in[1];
  const float* w1  = (const float*)d_in[2];
  const float* b1  = (const float*)d_in[3];
  const float* w2  = (const float*)d_in[4];
  const float* b2  = (const float*)d_in[5];
  const float* g1  = (const float*)d_in[6];
  const float* be1 = (const float*)d_in[7];
  const float* g2  = (const float*)d_in[8];
  const float* be2 = (const float*)d_in[9];
  float* out = (float*)d_out;
  char*  ws  = (char*)d_ws;

  // Workspace layout (bytes)
  unsigned short* xb   = (unsigned short*)(ws + 0L);          // 33,554,432
  unsigned short* wpb  = (unsigned short*)(ws + 33554432L);   //  5,242,880
  unsigned short* w1b  = (unsigned short*)(ws + 38797312L);   //  2,097,152
  unsigned short* w2b  = (unsigned short*)(ws + 40894464L);   //  2,097,152
  unsigned short* qb   = (unsigned short*)(ws + 42991616L);   // 33,554,432
  unsigned short* kb   = (unsigned short*)(ws + 76546048L);   // 33,554,432
  unsigned short* vtb  = (unsigned short*)(ws + 110100480L);  // 16,777,216
  float*          attn = (float*)(ws + 126877696L);           // 33,554,432 (reused as y)
  float*          hF   = (float*)(ws + 160432128L);           // 33,554,432
  unsigned short* hB   = (unsigned short*)(ws + 193986560L);  // 16,777,216
  unsigned short* ffb  = (unsigned short*)(ws + 210763776L);  // 67,108,864

  // 1) bf16 conversions
  long n4;
  n4 = (long)BL * EE / 4;
  f32_to_bf16_k<<<(unsigned)((n4 + 255) / 256), 256, 0, stream>>>(x, xb, n4);
  n4 = (long)2560 * EE / 4;
  f32_to_bf16_k<<<(unsigned)((n4 + 255) / 256), 256, 0, stream>>>(wp, wpb, n4);
  n4 = (long)FFD * NDD / 4;
  f32_to_bf16_k<<<(unsigned)((n4 + 255) / 256), 256, 0, stream>>>(w1, w1b, n4);
  n4 = (long)NDD * FFD / 4;
  f32_to_bf16_k<<<(unsigned)((n4 + 255) / 256), 256, 0, stream>>>(w2, w2b, n4);

  // 2) QKV projection: proj = x @ in_proj_w^T  (q scaled, v transposed)
  gemm_bf16_wmma<EE, 0><<<dim3(BL / 64, 2560 / 128), 256, 0, stream>>>(
      xb, wpb, nullptr, nullptr, nullptr, qb, kb, vtb);

  // 3) Flash attention -> attn (fp32, [BL, 512])
  flash_attn_k<<<BB * (LL / 64), 128, FA_LDS_BYTES, stream>>>(qb, kb, vtb, attn);

  // 4) h = LN(attn) * g1 + be1  (fp32 + bf16 copies)
  ln512_k<<<BL / 8, 256, 0, stream>>>(attn, g1, be1, hF, hB);

  // 5) ff = relu(h @ w1^T + b1) -> bf16 [BL, 2048]
  gemm_bf16_wmma<NDD, 1><<<dim3(BL / 64, FFD / 128), 256, 0, stream>>>(
      hB, w1b, b1, nullptr, nullptr, ffb, nullptr, nullptr);

  // 6) y = ff @ w2^T + b2 + h  (fp32, reuse attn buffer)
  gemm_bf16_wmma<FFD, 2><<<dim3(BL / 64, NDD / 128), 256, 0, stream>>>(
      ffb, w2b, b2, hF, attn, nullptr, nullptr, nullptr);

  // 7) out = LN(y) * g2 + be2
  ln512_k<<<BL / 8, 256, 0, stream>>>(attn, g2, be2, out, nullptr);
}
